// Model_39676907882498
// MI455X (gfx1250) — matrix operand order repaired and checked by reading, not compiled, not run on hardware
//
#include <hip/hip_runtime.h>

typedef __attribute__((ext_vector_type(16))) _Float16 v16h;
typedef __attribute__((ext_vector_type(2)))  _Float16 h2;
typedef __attribute__((ext_vector_type(8)))  float    v8f;
typedef int v4i __attribute__((vector_size(16)));

#define S_LEN 2048
#define D_DIM 64

#define AS1 __attribute__((address_space(1)))
#define AS3 __attribute__((address_space(3)))

#if defined(__has_builtin)
#if __has_builtin(__builtin_amdgcn_global_load_async_to_lds_b128)
#define HAVE_ASYNC_LDS 1
#endif
#endif

#if defined(__has_builtin) && __has_builtin(__builtin_amdgcn_s_wait_asynccnt)
#define WAIT_ASYNC() __builtin_amdgcn_s_wait_asynccnt(0)
#else
#define WAIT_ASYNC() asm volatile("s_wait_asynccnt 0x0" ::: "memory")
#endif

// Flat->addrspace pointer production via integer casts:
//  - global: flat address == as(1) address
//  - LDS: ISA 10.2 aperture mapping puts the LDS offset in addr[31:0]
__device__ __forceinline__ AS1 v4i* as_global_v4(const void* p) {
    return (AS1 v4i*)(unsigned long long)(uintptr_t)p;
}
__device__ __forceinline__ AS3 v4i* as_lds_v4(const void* p) {
    return (AS3 v4i*)(unsigned)(uintptr_t)p;
}

// 16-lane xor-butterfly step. v_permlane16_b32 does an arbitrary gather within
// each 16-lane row: sel nibble i = source lane for lane i ({hi,lo} = 64-bit sel).
__device__ __forceinline__ float lane16_xor(float x, unsigned lo, unsigned hi, int off) {
#if __has_builtin(__builtin_amdgcn_permlane16)
    int v = __float_as_int(x);
    int t = __builtin_amdgcn_permlane16(v, v, lo, hi, false, false);
    return __int_as_float(t);
#else
    return __shfl_xor(x, off, 32);
#endif
}

// Flash attention forward, fp32 in/out, f16 WMMA with f32 accumulation.
// Block = 128 threads (4 waves). Each wave owns a 16-row Q tile; the block
// processes 64 query rows and streams K/V in 64-row tiles through LDS.
// K tiles arrive one iteration ahead via GLOBAL_LOAD_ASYNC_TO_LDS (ASYNCcnt),
// double-buffered in raw f32, then packed to f16 in a cheap LDS->LDS pass.
__global__ __launch_bounds__(128) void fa_fwd_kernel(const float* __restrict__ Q,
                                                     const float* __restrict__ K,
                                                     const float* __restrict__ V,
                                                     const float* __restrict__ inv_scale,
                                                     float* __restrict__ O)
{
#ifdef HAVE_ASYNC_LDS
    __shared__ __align__(16) float    fK [2][64 * D_DIM]; // raw f32 K tiles (async dst)
#endif
    __shared__ __align__(32) _Float16 lK [64 * D_DIM];    // [kv][d] f16, row-major
    __shared__ __align__(32) _Float16 lVt[D_DIM * 64];    // [d][kv] f16, transposed
    __shared__ __align__(32) _Float16 lP [4][16 * 64];    // per-wave P tile [m][kv]

    const int tid  = threadIdx.x;
    const int lane = tid & 31;
    const int wave = tid >> 5;
    const int half = lane >> 4;   // which 16-lane group (A kgroup / C row-offset)
    const int l16  = lane & 15;

    const int qtiles = S_LEN / 64;
    const int qt = blockIdx.x % qtiles;
    const int bh = blockIdx.x / qtiles;   // fused (b,h): 0..31

    const float* Qb = Q + (size_t)bh * S_LEN * D_DIM;
    const float* Kb = K + (size_t)bh * S_LEN * D_DIM;
    const float* Vb = V + (size_t)bh * S_LEN * D_DIM;
    float*       Ob = O + (size_t)bh * S_LEN * D_DIM;

    const int   q0    = qt * 64 + wave * 16;       // this wave's first q row
    const float scale = 1.0f / inv_scale[0];       // 1/sqrt(D)

    // ---- Q fragment as WMMA A operand (16-bit A 16x32, contiguous k order matching B) ----
    // lane l16 = row M; element e -> K = half*16+e
    // Pre-multiply by 1/sqrt(D) here so S comes out of WMMA already scaled.
    v16h aQ0, aQ1;   // D slices 0..31 and 32..63
    {
        const float* qr = Qb + (size_t)(q0 + l16) * D_DIM;
        #pragma unroll
        for (int e = 0; e < 16; ++e) {
            const int k = half * 16 + e;
            aQ0[e] = (_Float16)(qr[k] * scale);
            aQ1[e] = (_Float16)(qr[32 + k] * scale);
        }
    }

    // All-ones B fragment: l-accumulator row-sum comes from a WMMA (P x 1).
    v16h bOnes;
    #pragma unroll
    for (int e = 0; e < 16; ++e) bOnes[e] = (_Float16)1.0f;

    // O accumulators: 4 N-blocks of 16 (D=64), C layout (lane=col, vgpr=row)
    v8f o0 = {}, o1 = {}, o2 = {}, o3 = {};
    v8f lacc = {};                 // row sums of exp(S-m), via ones-WMMA
    float mrow[8];
    #pragma unroll
    for (int r = 0; r < 8; ++r) mrow[r] = -__builtin_inff();

    _Float16* pw = lP[wave];

#ifdef HAVE_ASYNC_LDS
    // Preload K tile 0 asynchronously (8 x b128 per thread = 16KB per tile).
    {
        const float* src = Kb;
        float*       dst = fK[0];
        #pragma unroll
        for (int s = 0; s < 8; ++s) {
            const int off = s * 512 + tid * 4;
            __builtin_amdgcn_global_load_async_to_lds_b128(as_global_v4(src + off),
                                                           as_lds_v4(dst + off), 0, 0);
        }
    }
#endif

    for (int kv = 0; kv < S_LEN; kv += 64) {
#ifdef HAVE_ASYNC_LDS
        WAIT_ASYNC();      // this tile's async K data has landed (own cnt)
#endif
        __syncthreads();   // ...and is visible; previous tile reads done

#ifdef HAVE_ASYNC_LDS
        const int buf = (kv >> 6) & 1;
        if (kv + 64 < S_LEN) {   // kick next K tile into the other buffer
            const float* src = Kb + (size_t)(kv + 64) * D_DIM;
            float*       dst = fK[buf ^ 1];
            #pragma unroll
            for (int s = 0; s < 8; ++s) {
                const int off = s * 512 + tid * 4;
                __builtin_amdgcn_global_load_async_to_lds_b128(as_global_v4(src + off),
                                                               as_lds_v4(dst + off), 0, 0);
            }
        }
        const float* fKcur = fK[buf];
#endif

        // ---- stage tiles as packed f16 (2 halfs per ds_store_b32) ----
        #pragma unroll
        for (int p = tid; p < 2048; p += 128) {
            // K tile row-major: pack two consecutive d
#ifdef HAVE_ASYNC_LDS
            const float2 kf = *(const float2*)&fKcur[2 * p];          // LDS f32
#else
            const int rk = p >> 5;          // kv row 0..63
            const int ck = (p & 31) * 2;    // d (even)
            const float2 kf = *(const float2*)&Kb[(size_t)(kv + rk) * D_DIM + ck];
#endif
            *(h2*)&lK[2 * p] = (h2){(_Float16)kf.x, (_Float16)kf.y};
            // V tile transposed [d][kv]: pack two consecutive kv rows
            const int cv = p & 63;          // d
            const int rv = (p >> 6) * 2;    // kv row (even)
            const float v0f = Vb[(size_t)(kv + rv    ) * D_DIM + cv];
            const float v1f = Vb[(size_t)(kv + rv + 1) * D_DIM + cv];
            *(h2*)&lVt[cv * 64 + rv] = (h2){(_Float16)v0f, (_Float16)v1f};
        }
        if (kv + 64 < S_LEN) {   // hint next V tile toward the caches
            __builtin_prefetch(&Vb[(size_t)(kv + 64) * D_DIM + tid * 8], 0, 1);
        }
        __syncthreads();

        // ---- S = Q * K^T, 4 col-blocks of 16 kv positions each ----
        // B fragment: lane = S-col n (kv row), K-dim = d slice.
        v8f s0 = {}, s1 = {}, s2 = {}, s3 = {};
        {
            const v16h b00 = *(const v16h*)&lK[(l16     ) * 64 +  0 + half * 16];
            const v16h b01 = *(const v16h*)&lK[(l16     ) * 64 + 32 + half * 16];
            const v16h b10 = *(const v16h*)&lK[(l16 + 16) * 64 +  0 + half * 16];
            const v16h b11 = *(const v16h*)&lK[(l16 + 16) * 64 + 32 + half * 16];
            const v16h b20 = *(const v16h*)&lK[(l16 + 32) * 64 +  0 + half * 16];
            const v16h b21 = *(const v16h*)&lK[(l16 + 32) * 64 + 32 + half * 16];
            const v16h b30 = *(const v16h*)&lK[(l16 + 48) * 64 +  0 + half * 16];
            const v16h b31 = *(const v16h*)&lK[(l16 + 48) * 64 + 32 + half * 16];
            s0 = __builtin_amdgcn_wmma_f32_16x16x32_f16(false, aQ0, false, b00, (short)0, s0, false, false);
            s0 = __builtin_amdgcn_wmma_f32_16x16x32_f16(false, aQ1, false, b01, (short)0, s0, false, false);
            s1 = __builtin_amdgcn_wmma_f32_16x16x32_f16(false, aQ0, false, b10, (short)0, s1, false, false);
            s1 = __builtin_amdgcn_wmma_f32_16x16x32_f16(false, aQ1, false, b11, (short)0, s1, false, false);
            s2 = __builtin_amdgcn_wmma_f32_16x16x32_f16(false, aQ0, false, b20, (short)0, s2, false, false);
            s2 = __builtin_amdgcn_wmma_f32_16x16x32_f16(false, aQ1, false, b21, (short)0, s2, false, false);
            s3 = __builtin_amdgcn_wmma_f32_16x16x32_f16(false, aQ0, false, b30, (short)0, s3, false, false);
            s3 = __builtin_amdgcn_wmma_f32_16x16x32_f16(false, aQ1, false, b31, (short)0, s3, false, false);
        }

        // ---- online softmax over 64 kv cols at once.
        // C layout: lane holds col l16 (+16j per block), rows r + 8*half.
        #pragma unroll
        for (int r = 0; r < 8; ++r) {
            const float a0 = s0[r], a1 = s1[r], a2 = s2[r], a3 = s3[r];
            float tm = fmaxf(fmaxf(a0, a1), fmaxf(a2, a3));
            tm = fmaxf(tm, lane16_xor(tm, 0x67452301u, 0xEFCDAB89u, 1));
            tm = fmaxf(tm, lane16_xor(tm, 0x54761032u, 0xDCFE98BAu, 2));
            tm = fmaxf(tm, lane16_xor(tm, 0x32107654u, 0xBA98FEDCu, 4));
            tm = fmaxf(tm, lane16_xor(tm, 0xFEDCBA98u, 0x76543210u, 8));
            const float mnew  = fmaxf(mrow[r], tm);
            const float alpha = __expf(mrow[r] - mnew);
            mrow[r] = mnew;
            const float p0 = __expf(a0 - mnew);
            const float p1 = __expf(a1 - mnew);
            const float p2 = __expf(a2 - mnew);
            const float p3 = __expf(a3 - mnew);
            lacc[r] *= alpha;
            o0[r] *= alpha; o1[r] *= alpha; o2[r] *= alpha; o3[r] *= alpha;
            const int m = r + 8 * half;
            pw[m * 64      + l16] = (_Float16)p0;   // wave-private LDS scratch
            pw[m * 64 + 16 + l16] = (_Float16)p1;
            pw[m * 64 + 32 + l16] = (_Float16)p2;
            pw[m * 64 + 48 + l16] = (_Float16)p3;
        }
        __builtin_amdgcn_wave_barrier();   // keep P stores before A-layout reads

        // ---- P reload in A-matrix layout: two 16x32 fragments ----
        v16h aP0, aP1;
        {
            const _Float16* pr = &pw[l16 * 64];
            #pragma unroll
            for (int e = 0; e < 16; ++e) {
                const int k = half * 16 + e;
                aP0[e] = pr[k];        // kv cols  0..31
                aP1[e] = pr[32 + k];   // kv cols 32..63
            }
        }

        // ---- l += rowsum(P) on the matrix core: P x ones ----
        lacc = __builtin_amdgcn_wmma_f32_16x16x32_f16(false, aP0, false, bOnes, (short)0, lacc, false, false);
        lacc = __builtin_amdgcn_wmma_f32_16x16x32_f16(false, aP1, false, bOnes, (short)0, lacc, false, false);

        // ---- O += P * V : B fragment lane = d-col, K-dim = kv (lVt contig) ----
        {
            const v16h c00 = *(const v16h*)&lVt[( 0 + l16) * 64 +  0 + half * 16];
            const v16h c01 = *(const v16h*)&lVt[( 0 + l16) * 64 + 32 + half * 16];
            const v16h c10 = *(const v16h*)&lVt[(16 + l16) * 64 +  0 + half * 16];
            const v16h c11 = *(const v16h*)&lVt[(16 + l16) * 64 + 32 + half * 16];
            const v16h c20 = *(const v16h*)&lVt[(32 + l16) * 64 +  0 + half * 16];
            const v16h c21 = *(const v16h*)&lVt[(32 + l16) * 64 + 32 + half * 16];
            const v16h c30 = *(const v16h*)&lVt[(48 + l16) * 64 +  0 + half * 16];
            const v16h c31 = *(const v16h*)&lVt[(48 + l16) * 64 + 32 + half * 16];
            o0 = __builtin_amdgcn_wmma_f32_16x16x32_f16(false, aP0, false, c00, (short)0, o0, false, false);
            o0 = __builtin_amdgcn_wmma_f32_16x16x32_f16(false, aP1, false, c01, (short)0, o0, false, false);
            o1 = __builtin_amdgcn_wmma_f32_16x16x32_f16(false, aP0, false, c10, (short)0, o1, false, false);
            o1 = __builtin_amdgcn_wmma_f32_16x16x32_f16(false, aP1, false, c11, (short)0, o1, false, false);
            o2 = __builtin_amdgcn_wmma_f32_16x16x32_f16(false, aP0, false, c20, (short)0, o2, false, false);
            o2 = __builtin_amdgcn_wmma_f32_16x16x32_f16(false, aP1, false, c21, (short)0, o2, false, false);
            o3 = __builtin_amdgcn_wmma_f32_16x16x32_f16(false, aP0, false, c30, (short)0, o3, false, false);
            o3 = __builtin_amdgcn_wmma_f32_16x16x32_f16(false, aP1, false, c31, (short)0, o3, false, false);
        }
    }

    // ---- normalize and store: lane = col l16, rows r + 8*half ----
    #pragma unroll
    for (int r = 0; r < 8; ++r) {
        const float invl = 1.0f / lacc[r];
        float* op = Ob + (size_t)(q0 + r + 8 * half) * D_DIM + l16;
        op[ 0] = o0[r] * invl;
        op[16] = o1[r] * invl;
        op[32] = o2[r] * invl;
        op[48] = o3[r] * invl;
    }
}

extern "C" void kernel_launch(void* const* d_in, const int* in_sizes, int n_in,
                              void* d_out, int out_size, void* d_ws, size_t ws_size,
                              hipStream_t stream) {
    const float* Q  = (const float*)d_in[0];
    const float* K  = (const float*)d_in[1];
    const float* V  = (const float*)d_in[2];
    const float* sc = (const float*)d_in[3];
    float* O        = (float*)d_out;

    // grid = B*H * (S/64) = 32 * 32 = 1024 blocks of 128 threads (4 waves)
    const int blocks = 2 * 16 * (S_LEN / 64);
    fa_fwd_kernel<<<blocks, 128, 0, stream>>>(Q, K, V, sc, O);
}
